// get_model_42228118454701
// MI455X (gfx1250) — matrix operand order repaired and checked by reading, not compiled, not run on hardware
//
#include <hip/hip_runtime.h>
#include <hip/hip_bf16.h>

typedef __attribute__((ext_vector_type(16))) _Float16 v16h;
typedef __attribute__((ext_vector_type(8)))  float    v8f;

struct alignas(16) U4 { unsigned int x[4]; };

static __device__ inline v16h ld_frag(const _Float16* p0, const _Float16* p1) {
  union { v16h v; U4 u[2]; } r;
  r.u[0] = *(const U4*)p0;
  r.u[1] = *(const U4*)p1;
  return r.v;
}

#define WMMA(a,b,c) __builtin_amdgcn_wmma_f32_16x16x32_f16(false,(a),false,(b),(short)0,(c),false,false)

// ---------- prep: convert + swizzle weights into B-fragment lane order ----------
// layout: [ntile][ktile(8)][lane(32)][16 halves], halves = 16 contiguous K,
// lane 0-15 -> K base 0, lane 16-31 -> K base 16 (contiguous K order, same as the A loads).
__global__ void prep_wswz(const float* __restrict__ w, _Float16* __restrict__ out, int total) {
  int o = blockIdx.x * 256 + threadIdx.x;
  if (o >= total) return;
  int e  = o & 15;
  int L  = (o >> 4) & 31;
  int kt = (o >> 9) & 7;
  int nt = o >> 12;
  int n  = nt * 16 + (L & 15);
  int k  = kt * 32 + ((L >> 4) << 4) + e;
  out[o] = (_Float16)w[n * 256 + k];
}

// bias pre-gathered into C-fragment order: [h][mt][nt][lane(32)][8 floats]
__global__ void prep_bias(const float* __restrict__ rpb, float* __restrict__ out) {
  int o = blockIdx.x * 256 + threadIdx.x; // total 8*4*4*32*8 = 32768
  if (o >= 32768) return;
  int vv = o & 7;
  int L  = (o >> 3) & 31;
  int nt = (o >> 8) & 3;
  int mt = (o >> 10) & 3;
  int h  = o >> 12;
  int m = vv + ((L >> 4) << 3) + 16 * mt;
  int n = (L & 15) + 16 * nt;
  int r1 = m >> 3, c1 = m & 7, r2 = n >> 3, c2 = n & 7;
  int idx = (r1 - r2 + 7) * 15 + (c1 - c2 + 7);
  out[o] = rpb[idx * 8 + h];
}

// ---------- main: one block (8 waves) per 8x8 window ----------
__global__ __launch_bounds__(256)
void swin_window_attn(const float* __restrict__ x,
                      const float* __restrict__ gamma,
                      const float* __restrict__ beta,
                      const float* __restrict__ scale_p,
                      const _Float16* __restrict__ wqkv,   // 48 ntiles swizzled
                      const _Float16* __restrict__ wproj,  // 16 ntiles swizzled
                      const float* __restrict__ biasf,     // C-frag bias
                      float* __restrict__ out) {
  // LDS layout (bytes), total 245760:
  //   [0,      73728) : s_p  f16 [8][64][72]   (early life: s_x32 f32 [64][257])
  //   [73728, 107520) : s_xr f16 [64][264]     residual copy (whole kernel)
  //   [107520,141312) : s_xn f16 [64][264]     (late life: s_o, same size)
  //   [141312,175104) : s_q  f16 [64][264]     (late life: s_y f32 [64][257] spans q+k)
  //   [175104,208896) : s_k  f16 [64][264]
  //   [208896,245760) : s_vT f16 [256][72]
  __shared__ __align__(16) unsigned char smem[245760];
  float*    s_x32 = (float*)smem;
  _Float16* s_p   = (_Float16*)smem;
  _Float16* s_xr  = (_Float16*)(smem + 73728);
  _Float16* s_xn  = (_Float16*)(smem + 107520);
  _Float16* s_o   = s_xn;
  _Float16* s_q   = (_Float16*)(smem + 141312);
  _Float16* s_k   = (_Float16*)(smem + 175104);
  _Float16* s_vT  = (_Float16*)(smem + 208896);
  float*    s_y   = (float*)(smem + 141312);

  const int tid  = threadIdx.x;
  const int lane = tid & 31;
  const int wid  = tid >> 5;
  const int win  = blockIdx.x;
  const int b    = win >> 8;
  const int wh   = (win >> 4) & 15;
  const int ww   = win & 15;
  const int row0 = wh << 3, col0 = ww << 3;
  const long long imgBase = (long long)b * 256 * 128 * 128;

  const int half16 = (lane >> 4) << 3;   // C-frag row sub-base (rows)
  const int bsub   = (lane >> 4) << 4;   // B-frag K sub-base (halves)
  const int ln15   = lane & 15;

  // ---- phase 0: load x window (coalesced), stash f32 for LN + f16 residual ----
  for (int i = tid; i < 64 * 256; i += 256) {
    int c = i >> 6, t = i & 63, r = t >> 3, cc = t & 7;
    float v = x[imgBase + ((long long)c * 128 + row0 + r) * 128 + col0 + cc];
    s_x32[t * 257 + c] = v;
    s_xr [t * 264 + c] = (_Float16)v;
  }
  __syncthreads();

  // ---- phase 1: LayerNorm (4 threads per token) ----
  {
    int t = tid >> 2, part = tid & 3;
    float sum = 0.f, sq = 0.f;
    for (int c = part; c < 256; c += 4) {
      float v = s_x32[t * 257 + c];
      sum += v; sq += v * v;
    }
    sum += __shfl_xor(sum, 1); sum += __shfl_xor(sum, 2);
    sq  += __shfl_xor(sq,  1); sq  += __shfl_xor(sq,  2);
    float mu   = sum * (1.f / 256.f);
    float var  = sq * (1.f / 256.f) - mu * mu;
    float rstd = rsqrtf(var + 1e-5f);
    for (int c = part; c < 256; c += 4) {
      float v = (s_x32[t * 257 + c] - mu) * rstd * gamma[c] + beta[c];
      s_xn[t * 264 + c] = (_Float16)v;
    }
  }
  __syncthreads();

  // ---- phase 2: QKV GEMM (64x256 @ 256x768), 6 N-tiles per wave ----
  for (int nn = 0; nn < 6; ++nn) {
    int nt = wid * 6 + nn;
    v8f acc[4] = {};
#pragma unroll
    for (int kt = 0; kt < 8; ++kt) {
      const _Float16* bp = wqkv + ((nt * 8 + kt) * 32 + lane) * 16;
      v16h bf = ld_frag(bp, bp + 8);
#pragma unroll
      for (int mt = 0; mt < 4; ++mt) {
        const _Float16* ap = s_xn + (ln15 + 16 * mt) * 264 + kt * 32 + bsub;
        v16h af = ld_frag(ap, ap + 8);
        acc[mt] = WMMA(af, bf, acc[mt]);
      }
    }
#pragma unroll
    for (int mt = 0; mt < 4; ++mt)
#pragma unroll
      for (int vv = 0; vv < 8; ++vv) {
        int m  = vv + half16 + 16 * mt;
        int ch = nt * 16 + ln15;
        float f = acc[mt][vv];
        if (ch < 256)      s_q[m * 264 + ch]            = (_Float16)(f * 0.17677669529663687f);
        else if (ch < 512) s_k[m * 264 + (ch - 256)]    = (_Float16)f;
        else               s_vT[(ch - 512) * 72 + m]    = (_Float16)f;
      }
  }
  __syncthreads();

  // ---- phase 3+4: one head per wave: scores + bias (as WMMA C), softmax, PV ----
  {
    int h = wid;
    _Float16* ph = s_p + h * 64 * 72;
#pragma unroll
    for (int mt = 0; mt < 4; ++mt) {
      const _Float16* ap = s_q + (ln15 + 16 * mt) * 264 + h * 32 + bsub;
      v16h qa = ld_frag(ap, ap + 8);
      v8f sc[4];
#pragma unroll
      for (int nt = 0; nt < 4; ++nt) {
        const _Float16* bp = s_k + (ln15 + 16 * nt) * 264 + h * 32 + bsub;
        v16h kb = ld_frag(bp, bp + 8);
        v8f cb = *(const v8f*)(biasf + (((h * 4 + mt) * 4 + nt) * 32 + lane) * 8);
        sc[nt] = WMMA(qa, kb, cb);
      }
#pragma unroll
      for (int vv = 0; vv < 8; ++vv) {
        float mx = fmaxf(fmaxf(sc[0][vv], sc[1][vv]), fmaxf(sc[2][vv], sc[3][vv]));
        for (int o = 1; o < 16; o <<= 1) mx = fmaxf(mx, __shfl_xor(mx, o, 16));
        float e0 = __expf(sc[0][vv] - mx);
        float e1 = __expf(sc[1][vv] - mx);
        float e2 = __expf(sc[2][vv] - mx);
        float e3 = __expf(sc[3][vv] - mx);
        float s = e0 + e1 + e2 + e3;
        for (int o = 1; o < 16; o <<= 1) s += __shfl_xor(s, o, 16);
        float inv = 1.0f / s;
        int m2 = vv + half16 + 16 * mt;
        ph[m2 * 72 + ln15     ] = (_Float16)(e0 * inv);
        ph[m2 * 72 + ln15 + 16] = (_Float16)(e1 * inv);
        ph[m2 * 72 + ln15 + 32] = (_Float16)(e2 * inv);
        ph[m2 * 72 + ln15 + 48] = (_Float16)(e3 * inv);
      }
    }
    // PV: P(64x64) @ V(64x32)
#pragma unroll
    for (int mt = 0; mt < 4; ++mt) {
      v8f oa[2] = {};
#pragma unroll
      for (int kt = 0; kt < 2; ++kt) {
        const _Float16* ap = ph + (ln15 + 16 * mt) * 72 + kt * 32 + bsub;
        v16h pa = ld_frag(ap, ap + 8);
#pragma unroll
        for (int nt = 0; nt < 2; ++nt) {
          const _Float16* bp = s_vT + (h * 32 + nt * 16 + ln15) * 72 + kt * 32 + bsub;
          v16h vb = ld_frag(bp, bp + 8);
          oa[nt] = WMMA(pa, vb, oa[nt]);
        }
      }
#pragma unroll
      for (int nt = 0; nt < 2; ++nt) {
#pragma unroll
        for (int vv = 0; vv < 8; ++vv) {
          int m2 = vv + half16 + 16 * mt;
          s_o[m2 * 264 + h * 32 + nt * 16 + ln15] = (_Float16)oa[nt][vv];
        }
      }
    }
  }
  __syncthreads();

  // ---- phase 5: proj GEMM (64x256 @ 256x256), 2 N-tiles per wave ----
  for (int nn = 0; nn < 2; ++nn) {
    int nt = wid * 2 + nn;
    v8f acc[4] = {};
#pragma unroll
    for (int kt = 0; kt < 8; ++kt) {
      const _Float16* bp = wproj + ((nt * 8 + kt) * 32 + lane) * 16;
      v16h bf = ld_frag(bp, bp + 8);
#pragma unroll
      for (int mt = 0; mt < 4; ++mt) {
        const _Float16* ap = s_o + (ln15 + 16 * mt) * 264 + kt * 32 + bsub;
        v16h af = ld_frag(ap, ap + 8);
        acc[mt] = WMMA(af, bf, acc[mt]);
      }
    }
#pragma unroll
    for (int mt = 0; mt < 4; ++mt) {
#pragma unroll
      for (int vv = 0; vv < 8; ++vv) {
        int m = vv + half16 + 16 * mt;
        s_y[m * 257 + nt * 16 + ln15] = acc[mt][vv];
      }
    }
  }
  __syncthreads();

  // ---- epilogue: residual + scale, coalesced store ----
  float scale = scale_p[0];
  for (int i = tid; i < 64 * 256; i += 256) {
    int c = i >> 6, t = i & 63, r = t >> 3, cc = t & 7;
    float res = (float)s_xr[t * 264 + c] + scale * s_y[t * 257 + c];
    out[imgBase + ((long long)c * 128 + row0 + r) * 128 + col0 + cc] = res;
  }
}

extern "C" void kernel_launch(void* const* d_in, const int* in_sizes, int n_in,
                              void* d_out, int out_size, void* d_ws, size_t ws_size,
                              hipStream_t stream) {
  const float* x      = (const float*)d_in[0];
  const float* gamma  = (const float*)d_in[1];
  const float* beta   = (const float*)d_in[2];
  const float* w_qkv  = (const float*)d_in[3];
  const float* w_proj = (const float*)d_in[4];
  const float* scale  = (const float*)d_in[5];
  const float* rpb    = (const float*)d_in[6];
  float* out = (float*)d_out;

  char* wsb = (char*)d_ws;
  _Float16* wqkv_swz  = (_Float16*)(wsb);            // 768*256*2  = 393216 B
  _Float16* wproj_swz = (_Float16*)(wsb + 393216);   // 256*256*2  = 131072 B
  float*    bias_swz  = (float*)   (wsb + 524288);   // 32768*4    = 131072 B

  prep_wswz<<<768, 256, 0, stream>>>(w_qkv,  wqkv_swz,  48 * 8 * 32 * 16);
  prep_wswz<<<256, 256, 0, stream>>>(w_proj, wproj_swz, 16 * 8 * 32 * 16);
  prep_bias<<<128, 256, 0, stream>>>(rpb, bias_swz);

  swin_window_attn<<<4096, 256, 0, stream>>>(x, gamma, beta, scale,
                                             wqkv_swz, wproj_swz, bias_swz, out);
}
